// PVTAttention_3246995276308
// MI455X (gfx1250) — compile-verified, matrix operand order checked by reading, not run on hardware
//
#include <hip/hip_runtime.h>
#include <math.h>

#define CDIM 256
#define NTOT 12544
#define HEADS 8
#define HD 32
#define MKEY 49

typedef __bf16 bf16_t;
typedef __attribute__((ext_vector_type(16))) __bf16 v16bf;
typedef __attribute__((ext_vector_type(8)))  float  v8f;

// ---------------------------------------------------------------------------
// Tiled 256-deep GEMM:  Out[b][oc][n] = sum_ic W[oc][ic] * In[b][ic][n]
// 128x64 tile per block, 4 waves; each wave: 32 rows x 64 cols
// (2 A-fragments x 4 B-fragments = 8 WMMAs per 32-deep K step)
// ---------------------------------------------------------------------------
__global__ __launch_bounds__(128) void gemm256_kernel(
    const float* __restrict__ In, const float* __restrict__ W,
    float* __restrict__ Out)
{
  const int ntile  = blockIdx.x;
  const int octile = blockIdx.y;       // 0..1, 128 rows each
  const int b      = blockIdx.z;
  const int t      = threadIdx.x;
  const int wave   = t >> 5;
  const int lane   = t & 31;
  const int ml     = lane & 15;
  const int g      = lane >> 4;

  __shared__ bf16_t As[128 * 32];      // [oc_local][k]
  __shared__ bf16_t Bs[64 * 32];       // transposed: [n_local][k]

  const int oc_base = octile * 128;
  const int n_base  = ntile * 64;
  const float* inb  = In + (size_t)b * CDIM * NTOT;

  v8f acc[2][4];
  const v8f vzero = {0.f,0.f,0.f,0.f,0.f,0.f,0.f,0.f};
  #pragma unroll
  for (int ai = 0; ai < 2; ++ai)
    #pragma unroll
    for (int nt = 0; nt < 4; ++nt) acc[ai][nt] = vzero;

  for (int k0 = 0; k0 < CDIM; k0 += 32) {
    // stage A tile (weights): thread t loads one full 32-wide row
    {
      const float* src = W + (size_t)(oc_base + t) * CDIM + k0;
      bf16_t* dst = &As[t * 32];
      #pragma unroll
      for (int i = 0; i < 16; ++i) dst[i] = (bf16_t)src[i];
      #pragma unroll
      for (int i = 16; i < 32; ++i) dst[i] = (bf16_t)src[i];
    }
    // stage B tile (activations) transposed: [n][k], 16 elems/thread
    {
      const int kk = t >> 2;
      const int c0 = (t & 3) * 16;
      const float* src = inb + (size_t)(k0 + kk) * NTOT + n_base + c0;
      #pragma unroll
      for (int i = 0; i < 16; ++i) Bs[(c0 + i) * 32 + kk] = (bf16_t)src[i];
      if (k0 + 32 < CDIM) {
        __builtin_prefetch(src + 32 * NTOT, 0, 1);   // global_prefetch_b8
      }
    }
    __syncthreads();

    // load ALL fragments first, then issue the 8 WMMAs back-to-back
    v16bf a[2], bb[4];
    #pragma unroll
    for (int ai = 0; ai < 2; ++ai) {
      const bf16_t* ap = &As[(wave * 32 + ai * 16 + ml) * 32];
      #pragma unroll
      for (int j = 0; j < 8; ++j) { a[ai][j] = ap[8*g + j]; a[ai][8 + j] = ap[16 + 8*g + j]; }
    }
    #pragma unroll
    for (int nt = 0; nt < 4; ++nt) {
      const bf16_t* bp = &Bs[(nt * 16 + ml) * 32];
      #pragma unroll
      for (int j = 0; j < 8; ++j) { bb[nt][j] = bp[8*g + j]; bb[nt][8 + j] = bp[16 + 8*g + j]; }
    }
    #pragma unroll
    for (int ai = 0; ai < 2; ++ai)
      #pragma unroll
      for (int nt = 0; nt < 4; ++nt)
        acc[ai][nt] = __builtin_amdgcn_wmma_f32_16x16x32_bf16(
            false, a[ai], false, bb[nt], (short)0, acc[ai][nt], false, false);
    __syncthreads();
  }

  #pragma unroll
  for (int ai = 0; ai < 2; ++ai)
    #pragma unroll
    for (int nt = 0; nt < 4; ++nt) {
      const int col = n_base + nt * 16 + ml;
      #pragma unroll
      for (int r = 0; r < 8; ++r) {
        const int row = oc_base + wave * 32 + ai * 16 + r + 8 * g;
        Out[(size_t)b * CDIM * NTOT + (size_t)row * NTOT + col] = acc[ai][nt][r];
      }
    }
}

// ---------------------------------------------------------------------------
// Depthwise 3x3, pad 1, full resolution
// ---------------------------------------------------------------------------
__global__ void dwconv3_kernel(const float* __restrict__ in,
                               const float* __restrict__ w3,
                               float* __restrict__ out,
                               int C, int Hh, int Ww, long total)
{
  long idx = (long)blockIdx.x * blockDim.x + threadIdx.x;
  if (idx >= total) return;
  int xw = idx % Ww;
  int yh = (idx / Ww) % Hh;
  int c  = (idx / ((long)Ww * Hh)) % C;
  long bc = idx / ((long)Ww * Hh);
  const float* src = in + bc * Hh * Ww;
  const float* wc  = w3 + c * 9;
  float acc = 0.f;
  #pragma unroll
  for (int dy = 0; dy < 3; ++dy) {
    int yy = yh + dy - 1;
    if (yy < 0 || yy >= Hh) continue;
    #pragma unroll
    for (int dx = 0; dx < 3; ++dx) {
      int xx = xw + dx - 1;
      if (xx < 0 || xx >= Ww) continue;
      acc += src[yy * Ww + xx] * wc[dy * 3 + dx];
    }
  }
  out[idx] = acc;
}

// ---------------------------------------------------------------------------
// L2-normalize q per (b,head,n) over d=32; emit bf16 Q[bh][n][32]
// ---------------------------------------------------------------------------
__global__ void qnorm_kernel(const float* __restrict__ q, bf16_t* __restrict__ Qb)
{
  long idx = (long)blockIdx.x * blockDim.x + threadIdx.x;
  if (idx >= (long)8 * HEADS * NTOT) return;
  int n    = idx % NTOT;
  int head = (idx / NTOT) % HEADS;
  int b    = idx / ((long)NTOT * HEADS);
  const float* base = q + ((size_t)(b * CDIM + head * HD)) * NTOT + n;
  float v[HD];
  float ss = 0.f;
  #pragma unroll
  for (int d = 0; d < HD; ++d) { v[d] = base[(size_t)d * NTOT]; ss += v[d] * v[d]; }
  float inv = 1.f / fmaxf(sqrtf(ss), 1e-12f);
  bf16_t* dst = Qb + ((size_t)(b * HEADS + head) * NTOT + n) * HD;
  #pragma unroll
  for (int d = 0; d < HD; ++d) dst[d] = (bf16_t)(v[d] * inv);
}

// ---------------------------------------------------------------------------
// 16x16 average pool -> xp[b][c][49]
// ---------------------------------------------------------------------------
__global__ void avgpool_kernel(const float* __restrict__ x, float* __restrict__ xp)
{
  long idx = (long)blockIdx.x * blockDim.x + threadIdx.x;
  if (idx >= (long)8 * CDIM * 49) return;
  int p = idx % 49;
  long bc = idx / 49;
  int a1 = p / 7, a2 = p % 7;
  const float* src = x + (bc * 112 + a1 * 16) * 112 + a2 * 16;
  float s = 0.f;
  for (int i = 0; i < 16; ++i)
    #pragma unroll
    for (int j = 0; j < 16; ++j) s += src[i * 112 + j];
  xp[idx] = s * (1.f / 256.f);
}

// ---------------------------------------------------------------------------
// sr conv1x1 + channel LayerNorm + exact GELU.  One block per (b, m).
// ---------------------------------------------------------------------------
__global__ __launch_bounds__(256) void srln_kernel(
    const float* __restrict__ xp, const float* __restrict__ sr_w,
    const float* __restrict__ sr_b, const float* __restrict__ ln_w,
    const float* __restrict__ ln_b, float* __restrict__ xo)
{
  const int b  = blockIdx.x / 49;
  const int mm = blockIdx.x % 49;
  const int t  = threadIdx.x;
  const float* xpc = xp + (size_t)b * CDIM * 49 + mm;
  const float* wr  = sr_w + (size_t)t * CDIM;
  float acc = sr_b[t];
  for (int ic = 0; ic < CDIM; ++ic) acc += wr[ic] * xpc[(size_t)ic * 49];

  __shared__ float red[256];
  red[t] = acc; __syncthreads();
  for (int s = 128; s > 0; s >>= 1) { if (t < s) red[t] += red[t + s]; __syncthreads(); }
  float mu = red[0] * (1.f / 256.f);
  __syncthreads();
  float d = acc - mu;
  red[t] = d * d; __syncthreads();
  for (int s = 128; s > 0; s >>= 1) { if (t < s) red[t] += red[t + s]; __syncthreads(); }
  float var = red[0] * (1.f / 256.f);
  float xn = d * rsqrtf(var + 1e-5f) * ln_w[t] + ln_b[t];
  float ge = 0.5f * xn * (1.f + erff(xn * 0.70710678118654752f));
  xo[(size_t)b * CDIM * 49 + (size_t)t * 49 + mm] = ge;
}

// ---------------------------------------------------------------------------
// kv conv1x1 (256 -> 512 channels over 49 pixels)
// ---------------------------------------------------------------------------
__global__ void kvconv_kernel(const float* __restrict__ x_,
                              const float* __restrict__ kv_w,
                              float* __restrict__ kv)
{
  long idx = (long)blockIdx.x * blockDim.x + threadIdx.x;
  if (idx >= (long)8 * 512 * 49) return;
  int mm = idx % 49;
  int oc = (idx / 49) % 512;
  int b  = idx / ((long)49 * 512);
  const float* wr = kv_w + (size_t)oc * CDIM;
  const float* xc = x_ + (size_t)b * CDIM * 49 + mm;
  float acc = 0.f;
  for (int ic = 0; ic < CDIM; ++ic) acc += wr[ic] * xc[(size_t)ic * 49];
  kv[idx] = acc;
}

__global__ void zero_bf16_kernel(bf16_t* p, long n)
{
  long idx = (long)blockIdx.x * blockDim.x + threadIdx.x;
  if (idx < n) p[idx] = (bf16_t)0.f;
}

// ---------------------------------------------------------------------------
// kv depthwise 3x3 on 7x7; route to K[bh][64(pad m)][32] and Vt[bh][32][64]
// ---------------------------------------------------------------------------
__global__ void kvdw_kernel(const float* __restrict__ kv,
                            const float* __restrict__ w3,
                            bf16_t* __restrict__ Kpad, bf16_t* __restrict__ Vt)
{
  long idx = (long)blockIdx.x * blockDim.x + threadIdx.x;
  if (idx >= (long)8 * 512 * 49) return;
  int mm = idx % 49;
  int oc = (idx / 49) % 512;
  int b  = idx / ((long)49 * 512);
  int a1 = mm / 7, a2 = mm % 7;
  const float* src = kv + ((size_t)(b * 512 + oc)) * 49;
  const float* wc  = w3 + oc * 9;
  float acc = 0.f;
  #pragma unroll
  for (int dy = 0; dy < 3; ++dy) {
    int yy = a1 + dy - 1;
    if (yy < 0 || yy >= 7) continue;
    #pragma unroll
    for (int dx = 0; dx < 3; ++dx) {
      int xx = a2 + dx - 1;
      if (xx < 0 || xx >= 7) continue;
      acc += src[yy * 7 + xx] * wc[dy * 3 + dx];
    }
  }
  if (oc < CDIM) {
    int head = oc >> 5, d = oc & 31;
    Kpad[((size_t)(b * HEADS + head) * 64 + mm) * HD + d] = (bf16_t)acc;
  } else {
    int c = oc - CDIM;
    int head = c >> 5, d = c & 31;
    Vt[((size_t)(b * HEADS + head) * HD + d) * 64 + mm] = (bf16_t)acc;
  }
}

// ---------------------------------------------------------------------------
// Attention: per (b,head): S = Q K^T * scale, softmax over 49, O = P V.
// 4 waves/block, each wave owns 16 query rows. Out layout [b][c][n].
// ---------------------------------------------------------------------------
__global__ __launch_bounds__(128) void attn_kernel(
    const bf16_t* __restrict__ Qb, const bf16_t* __restrict__ Kpad,
    const bf16_t* __restrict__ Vt, const float* __restrict__ scale,
    float* __restrict__ Out)
{
  const int tile = blockIdx.x;            // 196 tiles of 64 query rows
  const int bh   = blockIdx.y;            // 64 = b*heads
  const int b    = bh >> 3, head = bh & 7;
  const int t    = threadIdx.x;
  const int wave = t >> 5, lane = t & 31;
  const int ml   = lane & 15, g = lane >> 4;

  __shared__ float  Ss[4][16 * 64];
  __shared__ bf16_t Ps[4][16 * 64];

  const int rowbase = tile * 64 + wave * 16;
  const float sc = scale[head];
  const v8f vzero = {0.f,0.f,0.f,0.f,0.f,0.f,0.f,0.f};

  // Q fragment: 16 rows x K=32 (full head dim in one WMMA)
  v16bf a;
  {
    const bf16_t* qp = Qb + ((size_t)bh * NTOT + rowbase + ml) * HD;
    #pragma unroll
    for (int j = 0; j < 8; ++j) { a[j] = qp[8*g + j]; a[8 + j] = qp[16 + 8*g + j]; }
  }

  // S = Q K^T   (49 keys padded to 64): load all 4 K fragments, then 4 WMMAs
  v16bf bk[4];
  #pragma unroll
  for (int nt = 0; nt < 4; ++nt) {
    const bf16_t* kp = Kpad + ((size_t)bh * 64 + nt * 16 + ml) * HD;
    #pragma unroll
    for (int j = 0; j < 8; ++j) { bk[nt][j] = kp[8*g + j]; bk[nt][8 + j] = kp[16 + 8*g + j]; }
  }
  v8f s[4];
  #pragma unroll
  for (int nt = 0; nt < 4; ++nt)
    s[nt] = __builtin_amdgcn_wmma_f32_16x16x32_bf16(
        false, a, false, bk[nt], (short)0, vzero, false, false);

  #pragma unroll
  for (int nt = 0; nt < 4; ++nt)
    #pragma unroll
    for (int r = 0; r < 8; ++r)
      Ss[wave][(r + 8 * g) * 64 + nt * 16 + ml] = s[nt][r] * sc;
  __syncthreads();

  // softmax over the 49 valid columns; lanes 0..15 each own a row
  if (lane < 16) {
    float* row = &Ss[wave][ml * 64];
    float mx = -1e30f;
    for (int n = 0; n < MKEY; ++n) mx = fmaxf(mx, row[n]);
    float sum = 0.f;
    for (int n = 0; n < MKEY; ++n) { float e = __expf(row[n] - mx); sum += e; row[n] = e; }
    float inv = 1.f / sum;
    bf16_t* prow = &Ps[wave][ml * 64];
    for (int n = 0; n < MKEY; ++n) prow[n] = (bf16_t)(row[n] * inv);
    for (int n = MKEY; n < 64; ++n) prow[n] = (bf16_t)0.f;
  }
  __syncthreads();

  // O = P (16x64) @ V (64x32): load all fragments, then 4 WMMAs
  v16bf ap[2], bv[2][2];
  #pragma unroll
  for (int ks = 0; ks < 2; ++ks) {
    const bf16_t* pp = &Ps[wave][ml * 64 + 32 * ks];
    #pragma unroll
    for (int j = 0; j < 8; ++j) { ap[ks][j] = pp[8*g + j]; ap[ks][8 + j] = pp[16 + 8*g + j]; }
    #pragma unroll
    for (int dt = 0; dt < 2; ++dt) {
      const bf16_t* vp = Vt + ((size_t)bh * HD + dt * 16 + ml) * 64 + 32 * ks;
      #pragma unroll
      for (int j = 0; j < 8; ++j) { bv[ks][dt][j] = vp[8*g + j]; bv[ks][dt][8 + j] = vp[16 + 8*g + j]; }
    }
  }
  v8f o[2];
  o[0] = vzero; o[1] = vzero;
  #pragma unroll
  for (int ks = 0; ks < 2; ++ks)
    #pragma unroll
    for (int dt = 0; dt < 2; ++dt)
      o[dt] = __builtin_amdgcn_wmma_f32_16x16x32_bf16(
          false, ap[ks], false, bv[ks][dt], (short)0, o[dt], false, false);

  #pragma unroll
  for (int dt = 0; dt < 2; ++dt) {
    const int d = dt * 16 + ml;
    #pragma unroll
    for (int r = 0; r < 8; ++r) {
      const int qrow = rowbase + r + 8 * g;
      Out[((size_t)b * CDIM + head * HD + d) * NTOT + qrow] = o[dt][r];
    }
  }
}

// ---------------------------------------------------------------------------
extern "C" void kernel_launch(void* const* d_in, const int* in_sizes, int n_in,
                              void* d_out, int out_size, void* d_ws, size_t ws_size,
                              hipStream_t stream)
{
  const float* x       = (const float*)d_in[0];
  const float* q_w     = (const float*)d_in[1];
  const float* q_dw_w  = (const float*)d_in[2];
  const float* kv_w    = (const float*)d_in[3];
  const float* kv_dw_w = (const float*)d_in[4];
  const float* sr_w    = (const float*)d_in[5];
  const float* sr_b    = (const float*)d_in[6];
  const float* ln_w    = (const float*)d_in[7];
  const float* ln_b    = (const float*)d_in[8];
  const float* scale   = (const float*)d_in[9];
  const float* proj_w  = (const float*)d_in[10];
  float* out = (float*)d_out;

  char* ws = (char*)d_ws;
  const size_t big = (size_t)8 * CDIM * NTOT;       // 25,690,112 elements
  float*  buf1  = (float*)(ws);                      // conv-q out, later attn out
  float*  buf2  = (float*)(ws + big * 4);            // dwconv-q out
  bf16_t* Qb    = (bf16_t*)(ws + big * 8);           // bf16 Q [bh][N][32]
  char*   small = ws + big * 8 + big * 2;
  float*  xp    = (float*)(small);                                   // 8*256*49
  float*  xmid  = (float*)(small + 8l * CDIM * 49 * 4);              // 8*256*49
  float*  kvbuf = (float*)(small + 16l * CDIM * 49 * 4);             // 8*512*49
  bf16_t* Kpad  = (bf16_t*)(small + 16l * CDIM * 49 * 4 + 8l * 512 * 49 * 4);
  bf16_t* Vt    = Kpad + (size_t)64 * 64 * HD;       // adjacent to Kpad

  // 1) q = conv1x1(x, q_w)  -> buf1        (WMMA GEMM)
  gemm256_kernel<<<dim3(NTOT / 64, CDIM / 128, 8), 128, 0, stream>>>(x, q_w, buf1);

  // 2) q = dwconv3(q)       -> buf2
  {
    long total = big;
    dwconv3_kernel<<<(total + 255) / 256, 256, 0, stream>>>(buf1, q_dw_w, buf2,
                                                            CDIM, 112, 112, total);
  }
  // 3) normalize q, pack bf16 -> Qb
  {
    long total = (long)8 * HEADS * NTOT;
    qnorm_kernel<<<(total + 255) / 256, 256, 0, stream>>>(buf2, Qb);
  }
  // 4) 16x16 avg pool -> xp
  {
    long total = (long)8 * CDIM * 49;
    avgpool_kernel<<<(total + 255) / 256, 256, 0, stream>>>(x, xp);
  }
  // 5) sr conv + LN + GELU -> xmid
  srln_kernel<<<8 * 49, 256, 0, stream>>>(xp, sr_w, sr_b, ln_w, ln_b, xmid);

  // 6) kv conv1x1 -> kvbuf
  {
    long total = (long)8 * 512 * 49;
    kvconv_kernel<<<(total + 255) / 256, 256, 0, stream>>>(xmid, kv_w, kvbuf);
  }
  // 7) zero K/V padding, then kv dwconv -> Kpad / Vt (bf16)
  {
    long nz = (long)64 * 64 * HD + (long)64 * HD * 64;
    zero_bf16_kernel<<<(nz + 255) / 256, 256, 0, stream>>>(Kpad, nz);
    long total = (long)8 * 512 * 49;
    kvdw_kernel<<<(total + 255) / 256, 256, 0, stream>>>(kvbuf, kv_dw_w, Kpad, Vt);
  }
  // 8) attention -> buf1 (reused)          (WMMA)
  attn_kernel<<<dim3(NTOT / 64, 8 * HEADS), 128, 0, stream>>>(Qb, Kpad, Vt, scale, buf1);

  // 9) proj conv1x1 -> out                 (WMMA GEMM)
  gemm256_kernel<<<dim3(NTOT / 64, CDIM / 128, 8), 128, 0, stream>>>(buf1, proj_w, out);
}
